// GeneReformer_28913719837488
// MI455X (gfx1250) — matrix operand order repaired and checked by reading, not compiled, not run on hardware
//
#include <hip/hip_runtime.h>

#define B_  8
#define S_  4096
#define H_  256
#define NH_ 4
#define D_  64
#define NHASH_ 2
#define CH_ 64
#define NC_ 64
#define NL_ 3
#define NBUCK_ 128

typedef __attribute__((ext_vector_type(16))) _Float16 v16h;
typedef __attribute__((ext_vector_type(8)))  _Float16 v8h;
typedef __attribute__((ext_vector_type(4)))  _Float16 v4h;
typedef __attribute__((ext_vector_type(8)))  float    v8f;

__device__ __forceinline__ v8f wmma16(v16h a, v16h b, v8f c) {
  return __builtin_amdgcn_wmma_f32_16x16x32_f16(false, a, false, b, (short)0, c, false, false);
}

// 16-bit A-matrix 16x32 fragment from row-major LDS [.., ld].
// lanes 0-15: M=lane, K in {0..15}; lanes 16-31: M=lane-16, K in {16..31}
// -> two contiguous 8-half runs per lane: force 2x ds_load_b128.
__device__ __forceinline__ v16h frag_a(const _Float16* p, int ld, int mbase, int kbase, int lane) {
  int row = mbase + (lane & 15);
  int kh  = (lane >> 4) * 16;
  const _Float16* base = p + row * ld + kbase + kh;
  v8h lo = *(const v8h*)(base);
  v8h hi = *(const v8h*)(base + 8);
  return __builtin_shufflevector(lo, hi, 0, 1, 2, 3, 4, 5, 6, 7,
                                         8, 9, 10, 11, 12, 13, 14, 15);
}

// B fragment from N-major storage (p[n][k], i.e. K^T source): 16 contiguous halves.
__device__ __forceinline__ v16h frag_bt(const _Float16* p, int ld, int kbase, int nbase, int lane) {
  int col  = nbase + (lane & 15);
  int koff = kbase + (lane >> 4) * 16;
  return *(const v16h*)(p + col * ld + koff);
}

// B fragment from fragment-ready swizzled LDS: tile index * 512 + lane*16, contiguous.
__device__ __forceinline__ v16h frag_b_swz(const _Float16* p, int tile, int lane) {
  return *(const v16h*)(p + tile * 512 + lane * 16);
}

// swizzled destination index for a B-tile element (k in [0,32), c in [0,64))
__device__ __forceinline__ int bswz_idx(int k, int c) {
  int ntile = c >> 4;
  int lane2 = (c & 15) | (((k >> 4) & 1) << 4);
  return ntile * 512 + lane2 * 16 + (k & 15);
}

// A fragment of softmax probabilities built on the fly: exp(score - z) -> f16
__device__ __forceinline__ v16h frag_p(const float* sS, const float* zrow, int mbase, int kbase, int lane) {
  int row = mbase + (lane & 15);
  int kh  = (lane >> 4) * 16;
  float zz = zrow[row];
  v16h a;
#pragma unroll
  for (int i = 0; i < 8; ++i) {
    int k = kbase + kh + 2 * i;
    a[2 * i]     = (_Float16)__expf(sS[row * 128 + k] - zz);
    a[2 * i + 1] = (_Float16)__expf(sS[row * 128 + k + 1] - zz);
  }
  return a;
}

// ---------------- generic WMMA GEMM: C = op(A[M,K] @ W[K,N] + bias) ----------------
// flags: 1 = relu, 2 = residual (C += result)
#define GBM 128
#define GBN 64
#define GBK 32

__global__ __launch_bounds__(256) void gemm_wmma(
    const float* __restrict__ A, const float* __restrict__ W,
    const float* __restrict__ bias, float* __restrict__ C,
    int M, int K, int N, int flags) {
  __shared__ __attribute__((aligned(32))) _Float16 sA[GBM * GBK];   // 8 KB row-major
  __shared__ __attribute__((aligned(32))) _Float16 sW[GBK * GBN];   // 4 KB swizzled
  int bm = blockIdx.x * GBM, bn = blockIdx.y * GBN;
  int tid = threadIdx.x, lane = tid & 31, wave = tid >> 5;
  v8f acc[4] = {};
  for (int k0 = 0; k0 < K; k0 += GBK) {
    // stage A: float4 global reads, 8-byte LDS stores
#pragma unroll
    for (int i = tid; i < GBM * GBK / 4; i += 256) {
      int r = i >> 3, c4 = (i & 7) * 4;
      int gr = bm + r;
      float4 f = make_float4(0.f, 0.f, 0.f, 0.f);
      if (gr < M) f = *(const float4*)(A + (size_t)gr * K + k0 + c4);
      v4h h = {(_Float16)f.x, (_Float16)f.y, (_Float16)f.z, (_Float16)f.w};
      *(v4h*)(sA + r * GBK + c4) = h;
    }
    // stage W swizzled into fragment-ready layout
#pragma unroll
    for (int i = tid; i < GBK * GBN / 4; i += 256) {
      int r = i >> 4, c4 = (i & 15) * 4;
      float4 f = *(const float4*)(W + (size_t)(k0 + r) * N + bn + c4);
      sW[bswz_idx(r, c4 + 0)] = (_Float16)f.x;
      sW[bswz_idx(r, c4 + 1)] = (_Float16)f.y;
      sW[bswz_idx(r, c4 + 2)] = (_Float16)f.z;
      sW[bswz_idx(r, c4 + 3)] = (_Float16)f.w;
    }
    __syncthreads();
    v16h af = frag_a(sA, GBK, wave * 16, 0, lane);
#pragma unroll
    for (int t = 0; t < 4; ++t) {
      v16h bf = frag_b_swz(sW, t, lane);
      acc[t] = wmma16(af, bf, acc[t]);
    }
    __syncthreads();
  }
  int mrow = bm + wave * 16 + ((lane >> 4) * 8);
#pragma unroll
  for (int t = 0; t < 4; ++t) {
    int col = bn + t * 16 + (lane & 15);
    float bv = bias ? bias[col] : 0.0f;
#pragma unroll
    for (int r = 0; r < 8; ++r) {
      int row = mrow + r;
      if (row < M) {
        float val = acc[t][r] + bv;
        if (flags & 1) val = fmaxf(val, 0.0f);
        size_t idx = (size_t)row * N + col;
        if (flags & 2) val += C[idx];
        C[idx] = val;
      }
    }
  }
}

// ---------------- embedding + concat: cat[bs, 0:256]=emb(pad0), [256:512]=expr linear ---
__global__ __launch_bounds__(256) void embed_kernel(
    const int* __restrict__ ids, const float* __restrict__ expr,
    const float* __restrict__ emb, const float* __restrict__ ew,
    const float* __restrict__ eb, float* __restrict__ cat) {
  size_t t = (size_t)blockIdx.x * 256 + threadIdx.x;   // < BS*512
  int col = (int)(t & 511);
  size_t bs = t >> 9;
  float v;
  if (col < H_) {
    int id = ids[bs];
    v = (id == 0) ? 0.0f : emb[(size_t)id * H_ + col];
  } else {
    int hc = col - H_;
    v = expr[bs] * ew[hc] + eb[hc];
  }
  cat[t] = v;
}

// ---------------- layer norm over 256-wide rows ----------------
__global__ __launch_bounds__(256) void ln256_kernel(
    const float* __restrict__ x, const float* __restrict__ g,
    const float* __restrict__ b, float* __restrict__ y, int relu) {
  __shared__ float red[256];
  int row = blockIdx.x, t = threadIdx.x;
  float v = x[(size_t)row * 256 + t];
  red[t] = v; __syncthreads();
  for (int s = 128; s > 0; s >>= 1) { if (t < s) red[t] += red[t + s]; __syncthreads(); }
  float mean = red[0] * (1.0f / 256.0f);
  __syncthreads();
  float d = v - mean;
  red[t] = d * d; __syncthreads();
  for (int s = 128; s > 0; s >>= 1) { if (t < s) red[t] += red[t + s]; __syncthreads(); }
  float var = red[0] * (1.0f / 256.0f);
  float o = d * rsqrtf(var + 1e-5f) * g[t] + b[t];
  if (relu) o = fmaxf(o, 0.0f);
  y[(size_t)row * 256 + t] = o;
}

// ---------------- k = q / (||q_head|| + 1e-6) ----------------
__global__ __launch_bounds__(64) void knorm_kernel(const float* __restrict__ q,
                                                   float* __restrict__ kk) {
  __shared__ float red[64];
  int bid = blockIdx.x;  // (b*S+s)*NH + h
  int t = threadIdx.x;
  size_t off = (size_t)(bid >> 2) * H_ + (size_t)(bid & 3) * D_;
  float v = q[off + t];
  red[t] = v * v; __syncthreads();
  for (int s = 32; s > 0; s >>= 1) { if (t < s) red[t] += red[t + s]; __syncthreads(); }
  float norm = sqrtf(red[0]);
  kk[off + t] = v / (norm + 1e-6f);
}

// ---------------- LSH hashing: bucket = argmax([r, -r]) ----------------
__global__ __launch_bounds__(64) void hash_kernel(const float* __restrict__ kk,
                                                  const float* __restrict__ rot,
                                                  int* __restrict__ buckets) {
  __shared__ float krow[64];
  __shared__ float rv[64];
  __shared__ int   ri[64];
  int bid = blockIdx.x;               // ((b*NH+h)*NHASH+n)*S + s (bit-packed)
  int s = bid & (S_ - 1);
  int n = (bid >> 12) & (NHASH_ - 1);
  int h = (bid >> 13) & (NH_ - 1);
  int b = bid >> 15;
  int t = threadIdx.x;
  krow[t] = kk[((size_t)b * S_ + s) * H_ + h * D_ + t];
  __syncthreads();
  float acc = 0.0f;
  for (int d = 0; d < D_; ++d)
    acc += krow[d] * rot[(((size_t)h * D_ + d) * NHASH_ + n) * (NBUCK_ / 2) + t];
  float v = acc; int idx = t;
  if (-acc > v) { v = -acc; idx = t + (NBUCK_ / 2); }
  rv[t] = v; ri[t] = idx;
  __syncthreads();
  for (int st = 32; st > 0; st >>= 1) {
    if (t < st) {
      if (rv[t + st] > rv[t] || (rv[t + st] == rv[t] && ri[t + st] < ri[t])) {
        rv[t] = rv[t + st]; ri[t] = ri[t + st];
      }
    }
    __syncthreads();
  }
  if (t == 0) buckets[(size_t)((b * NH_ + h) * NHASH_ + n) * S_ + s] = ri[0];
}

// ---------------- stable counting sort by bucket (== argsort(bucket*S + pos)) ----------
__global__ __launch_bounds__(128) void sort_kernel(const int* __restrict__ buckets,
                                                   int* __restrict__ sidx) {
  __shared__ int sb[S_];
  __shared__ int cnt[NBUCK_];
  __shared__ int off[NBUCK_ + 1];
  int bid = blockIdx.x, t = threadIdx.x;
  const int* bk = buckets + (size_t)bid * S_;
  for (int i = t; i < S_; i += 128) sb[i] = bk[i];
  __syncthreads();
  int c = 0;
  for (int i = 0; i < S_; ++i) c += (sb[i] == t);
  cnt[t] = c; __syncthreads();
  if (t == 0) { off[0] = 0; for (int i = 0; i < NBUCK_; ++i) off[i + 1] = off[i] + cnt[i]; }
  __syncthreads();
  int pos = off[t];
  int* so = sidx + (size_t)bid * S_;
  for (int i = 0; i < S_; ++i) if (sb[i] == t) so[pos++] = i;
}

// ---------------- fused chunk attention (per b,h,n,chunk) ----------------
__global__ __launch_bounds__(256) void attn_kernel(
    const float* __restrict__ q, const float* __restrict__ kk,
    const float* __restrict__ vv, const int* __restrict__ sidx,
    float* __restrict__ o, float* __restrict__ zg) {
  __shared__ __attribute__((aligned(64))) char smem[50176];
  _Float16* sQ = (_Float16*)smem;               // 64x64  f16 row-major (phase 1)
  _Float16* sK = (_Float16*)(smem + 8192);      // 128x64 f16 N-major   (phase 1)
  float*    sS = (float*)smem;                  // 64x128 f32 (phase 2, reuses sQ/sK)
  _Float16* sV = (_Float16*)(smem + 32768);     // 128x64 f16 swizzled fragment-ready
  int*   spos = (int*)(smem + 49152);           // 64
  int*   kpos = (int*)(smem + 49408);           // 128
  float* zrow = (float*)(smem + 49920);         // 64

  int bid = blockIdx.x;
  int c = bid & (NC_ - 1);
  int n = (bid >> 6) & (NHASH_ - 1);
  int h = (bid >> 8) & (NH_ - 1);
  int b = bid >> 10;
  int tid = threadIdx.x, lane = tid & 31, wave = tid >> 5;
  size_t base = (size_t)((b * NH_ + h) * NHASH_ + n) * S_;
  int pc = (c + NC_ - 1) & (NC_ - 1);   // roll by one chunk (wrap)

  if (tid < 64)       spos[tid]       = sidx[base + c  * CH_ + tid];
  else if (tid < 128) kpos[tid - 64]  = sidx[base + pc * CH_ + (tid - 64)];
  else if (tid < 192) kpos[tid - 64]  = sidx[base + c  * CH_ + (tid - 128)];
  __syncthreads();

  // stage Q (row-major), K (row-major, read later as K^T), V (swizzled)
#pragma unroll
  for (int i = tid; i < 64 * 64 / 4; i += 256) {
    int r = i >> 4, d4 = (i & 15) * 4;
    float4 f = *(const float4*)(q + ((size_t)b * S_ + spos[r]) * H_ + h * D_ + d4);
    v4h hq = {(_Float16)f.x, (_Float16)f.y, (_Float16)f.z, (_Float16)f.w};
    *(v4h*)(sQ + r * 64 + d4) = hq;
  }
#pragma unroll
  for (int i = tid; i < 128 * 64 / 4; i += 256) {
    int r = i >> 4, d4 = (i & 15) * 4;
    size_t src = ((size_t)b * S_ + kpos[r]) * H_ + h * D_ + d4;
    float4 fk = *(const float4*)(kk + src);
    v4h hk = {(_Float16)fk.x, (_Float16)fk.y, (_Float16)fk.z, (_Float16)fk.w};
    *(v4h*)(sK + r * 64 + d4) = hk;
    float4 fv = *(const float4*)(vv + src);
    int ks = r >> 5, kl = r & 31;
    _Float16* dstb = sV + (ks * 4) * 512;
    dstb[bswz_idx(kl, d4 + 0)] = (_Float16)fv.x;
    dstb[bswz_idx(kl, d4 + 1)] = (_Float16)fv.y;
    dstb[bswz_idx(kl, d4 + 2)] = (_Float16)fv.z;
    dstb[bswz_idx(kl, d4 + 3)] = (_Float16)fv.w;
  }
  __syncthreads();

  // GEMM1: scores[64][128] = Q @ K^T  (K=64 -> 2 wmma k-steps), 4 tiles/wave
  int mt  = wave >> 1;
  int ntb = (wave & 1) * 4;
  v8f a0 = {}, a1 = {}, a2 = {}, a3 = {};
#pragma unroll
  for (int ks = 0; ks < 2; ++ks) {
    v16h af = frag_a(sQ, 64, mt * 16, ks * 32, lane);
    a0 = wmma16(af, frag_bt(sK, 64, ks * 32, (ntb + 0) * 16, lane), a0);
    a1 = wmma16(af, frag_bt(sK, 64, ks * 32, (ntb + 1) * 16, lane), a1);
    a2 = wmma16(af, frag_bt(sK, 64, ks * 32, (ntb + 2) * 16, lane), a2);
    a3 = wmma16(af, frag_bt(sK, 64, ks * 32, (ntb + 3) * 16, lane), a3);
  }
  __syncthreads();  // all reads of sQ/sK done -> safe to overwrite with sS
  {
    int r0 = mt * 16 + ((lane >> 4) * 8);
    int c0 = lane & 15;
#pragma unroll
    for (int r = 0; r < 8; ++r) {
      sS[(r0 + r) * 128 + (ntb + 0) * 16 + c0] = a0[r] * 0.125f;
      sS[(r0 + r) * 128 + (ntb + 1) * 16 + c0] = a1[r] * 0.125f;
      sS[(r0 + r) * 128 + (ntb + 2) * 16 + c0] = a2[r] * 0.125f;
      sS[(r0 + r) * 128 + (ntb + 3) * 16 + c0] = a3[r] * 0.125f;
    }
  }
  __syncthreads();

  // self-mask + per-row logsumexp
  if (tid < 64) {
    int sp = spos[tid];
    float mx = -1e30f;
    for (int j = 0; j < 128; ++j) {
      float sv = sS[tid * 128 + j];
      if (kpos[j] == sp) sv -= 1e5f;
      sS[tid * 128 + j] = sv;
      mx = fmaxf(mx, sv);
    }
    float sum = 0.0f;
    for (int j = 0; j < 128; ++j) sum += __expf(sS[tid * 128 + j] - mx);
    float zz = mx + __logf(sum);
    zrow[tid] = zz;
    zg[base + sp] = zz;   // scatter to original token order
  }
  __syncthreads();

  // GEMM2: o[64][64] = P @ V  (K=128 -> 4 wmma k-steps), 2 tiles/wave
  int nt0 = (wave & 1) * 2;
  v8f o0 = {}, o1 = {};
#pragma unroll
  for (int ks = 0; ks < 4; ++ks) {
    v16h af = frag_p(sS, zrow, mt * 16, ks * 32, lane);
    o0 = wmma16(af, frag_b_swz(sV, ks * 4 + nt0 + 0, lane), o0);
    o1 = wmma16(af, frag_b_swz(sV, ks * 4 + nt0 + 1, lane), o1);
  }
  {
    int r0 = mt * 16 + ((lane >> 4) * 8);
    int c0 = lane & 15;
#pragma unroll
    for (int r = 0; r < 8; ++r) {
      int tok = spos[r0 + r];
      size_t ob = (base + tok) * D_;
      o[ob + (nt0 + 0) * 16 + c0] = o0[r];
      o[ob + (nt0 + 1) * 16 + c0] = o1[r];
    }
  }
}

// ---------------- combine hash rounds: softmax(z over n), weighted sum of o ----------
__global__ __launch_bounds__(256) void combine_kernel(const float* __restrict__ o,
                                                      const float* __restrict__ z,
                                                      float* __restrict__ out) {
  size_t t = (size_t)blockIdx.x * 256 + threadIdx.x;   // < B*S*H
  int d = (int)(t & 63);
  int h = (int)((t >> 6) & 3);
  int s = (int)((t >> 8) & (S_ - 1));
  int b = (int)(t >> 20);
  size_t zb = (size_t)((b * NH_ + h) * NHASH_) * S_ + s;
  float z0 = z[zb], z1 = z[zb + S_];
  float m = fmaxf(z0, z1);
  float w0 = __expf(z0 - m), w1 = __expf(z1 - m);
  float inv = 1.0f / (w0 + w1);
  size_t ob = ((size_t)((b * NH_ + h) * NHASH_) * S_ + s) * D_ + d;
  float val = (w0 * o[ob] + w1 * o[ob + (size_t)S_ * D_]) * inv;
  out[((size_t)b * S_ + s) * H_ + h * D_ + d] = val;
}

// ---------------- mean pool over sequence ----------------
__global__ __launch_bounds__(256) void pool_kernel(const float* __restrict__ x,
                                                   float* __restrict__ pooled) {
  int b = blockIdx.x, hc = threadIdx.x;
  float acc = 0.0f;
  for (int s = 0; s < S_; ++s) acc += x[((size_t)b * S_ + s) * H_ + hc];
  pooled[b * H_ + hc] = acc * (1.0f / S_);
}

extern "C" void kernel_launch(void* const* d_in, const int* in_sizes, int n_in,
                              void* d_out, int out_size, void* d_ws, size_t ws_size,
                              hipStream_t stream) {
  (void)in_sizes; (void)n_in; (void)out_size; (void)ws_size;
  const int*   gene_ids = (const int*)  d_in[0];
  const float* expr     = (const float*)d_in[1];
  const float* gene_emb = (const float*)d_in[2];
  const float* expr_w   = (const float*)d_in[3];
  const float* expr_b   = (const float*)d_in[4];
  const float* comb_w   = (const float*)d_in[5];
  const float* comb_b   = (const float*)d_in[6];
  const float* ln1_g    = (const float*)d_in[7];
  const float* ln1_b    = (const float*)d_in[8];
  const float* wqk      = (const float*)d_in[9];
  const float* wv       = (const float*)d_in[10];
  const float* wo       = (const float*)d_in[11];
  const float* wo_b     = (const float*)d_in[12];
  const float* ln2_g    = (const float*)d_in[13];
  const float* ln2_b    = (const float*)d_in[14];
  const float* w1       = (const float*)d_in[15];
  const float* b1       = (const float*)d_in[16];
  const float* w2       = (const float*)d_in[17];
  const float* b2       = (const float*)d_in[18];
  const float* rot      = (const float*)d_in[19];
  const float* out_w    = (const float*)d_in[20];
  const float* out_b    = (const float*)d_in[21];
  const float* lnf_g    = (const float*)d_in[22];
  const float* lnf_b    = (const float*)d_in[23];

  float* ws = (float*)d_ws;
  const size_t XSZ = (size_t)B_ * S_ * H_;        // 8.4M floats
  const size_t ZSZ = (size_t)B_ * NH_ * NHASH_ * S_;
  float* x    = ws;
  float* xn   = ws + 1 * XSZ;
  float* q    = ws + 2 * XSZ;
  float* kk   = ws + 3 * XSZ;
  float* vv   = ws + 4 * XSZ;
  float* o    = ws + 5 * XSZ;                     // 2*XSZ  [B,NH,NHASH,S,D]
  float* hid  = ws + 7 * XSZ;                     // 2*XSZ  (also concat buffer)
  float* z    = ws + 9 * XSZ;                     // ZSZ
  int*   buckets = (int*)(ws + 9 * XSZ + ZSZ);
  int*   sidx    = (int*)(ws + 9 * XSZ + 2 * ZSZ);
  float* pooled  = ws + 9 * XSZ + 3 * ZSZ;
  float* tmp     = pooled + (size_t)B_ * H_;

  const int BS = B_ * S_;
  dim3 blk256(256);

  embed_kernel<<<(BS * 2 * H_) / 256, blk256, 0, stream>>>(gene_ids, expr, gene_emb,
                                                           expr_w, expr_b, hid);
  gemm_wmma<<<dim3(BS / 128, H_ / 64), blk256, 0, stream>>>(hid, comb_w, comb_b, x,
                                                            BS, 2 * H_, H_, 0);

  for (int l = 0; l < NL_; ++l) {
    const float* wqk_l = wqk + (size_t)l * H_ * H_;
    const float* wv_l  = wv  + (size_t)l * H_ * H_;
    const float* wo_l  = wo  + (size_t)l * H_ * H_;
    const float* rot_l = rot + (size_t)l * NH_ * D_ * NHASH_ * (NBUCK_ / 2);

    ln256_kernel<<<BS, blk256, 0, stream>>>(x, ln1_g + l * H_, ln1_b + l * H_, xn, 0);
    gemm_wmma<<<dim3(BS / 128, H_ / 64), blk256, 0, stream>>>(xn, wqk_l, nullptr, q,
                                                              BS, H_, H_, 0);
    gemm_wmma<<<dim3(BS / 128, H_ / 64), blk256, 0, stream>>>(xn, wv_l, nullptr, vv,
                                                              BS, H_, H_, 0);
    knorm_kernel<<<BS * NH_, dim3(64), 0, stream>>>(q, kk);
    hash_kernel<<<B_ * NH_ * NHASH_ * S_, dim3(64), 0, stream>>>(kk, rot_l, buckets);
    sort_kernel<<<B_ * NH_ * NHASH_, dim3(128), 0, stream>>>(buckets, sidx);
    attn_kernel<<<B_ * NH_ * NHASH_ * NC_, blk256, 0, stream>>>(q, kk, vv, sidx, o, z);
    combine_kernel<<<(BS * H_) / 256, blk256, 0, stream>>>(o, z, xn);
    gemm_wmma<<<dim3(BS / 128, H_ / 64), blk256, 0, stream>>>(xn, wo_l, wo_b + l * H_, x,
                                                              BS, H_, H_, 2);
    ln256_kernel<<<BS, blk256, 0, stream>>>(x, ln2_g + l * H_, ln2_b + l * H_, xn, 0);
    gemm_wmma<<<dim3(BS / 128, (2 * H_) / 64), blk256, 0, stream>>>(
        xn, w1 + (size_t)l * H_ * 2 * H_, b1 + l * 2 * H_, hid, BS, H_, 2 * H_, 1);
    gemm_wmma<<<dim3(BS / 128, H_ / 64), blk256, 0, stream>>>(
        hid, w2 + (size_t)l * 2 * H_ * H_, b2 + l * H_, x, BS, 2 * H_, H_, 2);
  }

  pool_kernel<<<B_, blk256, 0, stream>>>(x, pooled);
  gemm_wmma<<<dim3(1, H_ / 64), blk256, 0, stream>>>(pooled, out_w, out_b, tmp,
                                                     B_, H_, H_, 0);
  ln256_kernel<<<B_, blk256, 0, stream>>>(tmp, lnf_g, lnf_b, (float*)d_out, 1);
}
